// LIFConvLayer_86509231276887
// MI455X (gfx1250) — matrix operand order repaired and checked by reading, not compiled, not run on hardware
//
#include <hip/hip_runtime.h>

// LIF recurrent layer, persistent-state formulation for MI455X (gfx1250).
// Each WG owns 16 batch rows for all 1024 steps; per-step [16x256]@[256x256]
// GEMM via v_wmma_f32_16x16x32_bf16. Weights bf16-transposed in LDS; one
// N-tile's B fragments register-resident; P' double-buffered (1 barrier/step).

typedef __attribute__((ext_vector_type(16))) __bf16 v16bf;
typedef __attribute__((ext_vector_type(8)))  __bf16 v8bf;
typedef __attribute__((ext_vector_type(2)))  __bf16 v2bf;
typedef __attribute__((ext_vector_type(8)))  float  v8f;
typedef __attribute__((ext_vector_type(4)))  float  v4f;

#define TT 1024
#define BB 512
#define CC 256
#define LDSROW 264            // 256 bf16 + 8 pad -> 528B stride, conflict-free b128 frag loads
#define PBUF   (16 * LDSROW)  // one P' buffer (bf16 elements)
#define THRV 1.0f

__device__ __forceinline__ __bf16 f2bf(float f) {
  unsigned u = __builtin_bit_cast(unsigned, f);
  u += 0x7FFFu + ((u >> 16) & 1u);              // round-to-nearest-even
  unsigned short h = (unsigned short)(u >> 16);
  return __builtin_bit_cast(__bf16, h);
}

__device__ __forceinline__ v2bf f2bf2(float a, float b) {
#if __has_builtin(__builtin_amdgcn_cvt_pk_bf16_f32)
  auto p = __builtin_amdgcn_cvt_pk_bf16_f32(a, b);   // v_cvt_pk_bf16_f32
  return __builtin_bit_cast(v2bf, p);
#else
  v2bf r; r[0] = f2bf(a); r[1] = f2bf(b); return r;
#endif
}

__device__ __forceinline__ v16bf joinbf(v8bf lo, v8bf hi) {
  v16bf r;
#pragma unroll
  for (int i = 0; i < 8; ++i) { r[i] = lo[i]; r[i + 8] = hi[i]; }
  return r;
}

extern "C" __global__ __launch_bounds__(256)
void lif_scan_kernel(const float* __restrict__ xin,  const float* __restrict__ w,
                     const float* __restrict__ bias, const float* __restrict__ alpha,
                     const float* __restrict__ beta, const float* __restrict__ gamma,
                     float* __restrict__ out) {
  extern __shared__ __align__(16) char smem_raw[];
  __bf16* wt   = (__bf16*)smem_raw;        // Wt[o][i] transposed bf16 weights: [CC][LDSROW]
  __bf16* plds = wt + CC * LDSROW;         // P' tiles, double buffered: 2 x [16][LDSROW]

  const int tid  = threadIdx.x;
  const int lane = tid & 31;
  const int wv   = tid >> 5;               // wave id 0..7 (wave32)
  const int rowbase = blockIdx.x * 16;

  // ---- one-time stage: Wt[o=tid][i] = bf16(W[i][o=tid])  (coalesced col gather)
  for (int i = 0; i < CC; i += 8) {
    v8bf pk;
#pragma unroll
    for (int j = 0; j < 4; ++j) {
      v2bf p2 = f2bf2(w[(size_t)(i + 2 * j) * CC + tid], w[(size_t)(i + 2 * j + 1) * CC + tid]);
      pk[2 * j] = p2[0]; pk[2 * j + 1] = p2[1];
    }
    *(v8bf*)(wt + (size_t)tid * LDSROW + i) = pk;
  }

  // ---- per-thread P/Q slice: row r, 16 consecutive channels at col0
  const int r    = tid >> 4;
  const int col0 = (tid & 15) * 16;
  float P[16], Q[16], al[16], be[16];
#pragma unroll
  for (int i = 0; i < 16; ++i) {
    P[i] = 0.f; Q[i] = 0.f;
    al[i] = alpha[col0 + i];
    be[i] = beta[col0 + i];
  }

  // ---- per-wave R/S state held in WMMA f32 accumulator layout, 2 N-tiles/wave
  const int n0  = wv * 2;
  const int nc  = lane & 15;               // N within tile
  const int mh  = (lane >> 4) & 1;         // acc VGPR i -> M = i + 8*mh
  const float g0 = gamma[n0 * 16 + nc];
  const float g1 = gamma[n0 * 16 + 16 + nc];
  const float b0 = bias [n0 * 16 + nc];
  const float b1 = bias [n0 * 16 + 16 + nc];
  v8f R0, R1, S0, S1;
#pragma unroll
  for (int i = 0; i < 8; ++i) { R0[i] = 0.f; R1[i] = 0.f; S0[i] = 0.f; S1[i] = 0.f; }

  // fragment addresses (bf16-element offsets), contiguous k order (k = 16h + s), same for A and B:
  // A lanes 0-15: K {0..15}; lanes 16-31: K {16..31}
  const int aoff = (lane & 15) * LDSROW + ((lane & 16) ? 16 : 0);
  const __bf16* abase0 = plds + aoff;
  const __bf16* abase1 = plds + PBUF + aoff;
  // B lanes 0-15: K {0..15}; lanes 16-31: K {16..31}  (row of Wt = fixed N)
  const int koffb = (lane & 16) ? 16 : 0;
  const __bf16* bbase0 = wt + (size_t)(n0 * 16 + nc)      * LDSROW + koffb;
  const __bf16* bbase1 = wt + (size_t)(n0 * 16 + 16 + nc) * LDSROW + koffb;

  const float* xptr = xin + ((size_t)rowbase + r) * CC + col0;
  float*       optr = out + (size_t)rowbase * CC;

  __syncthreads();                          // wt visible to all waves

  // ---- register-resident B fragments for first N-tile (loop-invariant)
  v16bf B0r[8];
#pragma unroll
  for (int k = 0; k < 8; ++k)
    B0r[k] = joinbf(*(const v8bf*)(bbase0 + k * 32),
                    *(const v8bf*)(bbase0 + k * 32 + 8));

  for (int t = 0; t < TT; ++t) {
    const __bf16* ab = (t & 1) ? abase1 : abase0;
    __bf16*       pw = (__bf16*)((t & 1) ? abase1 - aoff : abase0 - aoff) + r * LDSROW + col0;

    // stream in x_t slice (b128 loads) + prefetch next step
    const float* xp = xptr + (size_t)t * BB * CC;
    v4f x0 = *(const v4f*)(xp + 0);
    v4f x1 = *(const v4f*)(xp + 4);
    v4f x2 = *(const v4f*)(xp + 8);
    v4f x3 = *(const v4f*)(xp + 12);
    if (t + 1 < TT) __builtin_prefetch(xp + (size_t)BB * CC, 0, 3);

    float xv[16];
#pragma unroll
    for (int i = 0; i < 4; ++i) { xv[i] = x0[i]; xv[4+i] = x1[i]; xv[8+i] = x2[i]; xv[12+i] = x3[i]; }

    // P' = alpha*P + Q ; Q' = beta*Q + x_t   (old Q on RHS of P')
    float p2v[16];
#pragma unroll
    for (int i = 0; i < 16; ++i) {
      p2v[i] = fmaf(al[i], P[i], Q[i]);
      Q[i]   = fmaf(be[i], Q[i], xv[i]);
      P[i]   = p2v[i];
    }
    v8bf pk0, pk1;
#pragma unroll
    for (int j = 0; j < 4; ++j) {
      v2bf a2 = f2bf2(p2v[2 * j],     p2v[2 * j + 1]);
      v2bf c2 = f2bf2(p2v[2 * j + 8], p2v[2 * j + 9]);
      pk0[2 * j] = a2[0]; pk0[2 * j + 1] = a2[1];
      pk1[2 * j] = c2[0]; pk1[2 * j + 1] = c2[1];
    }
    *(v8bf*)(pw)     = pk0;
    *(v8bf*)(pw + 8) = pk1;

    __syncthreads();                        // P' buffer (t&1) ready; other buffer untouched

    // R' = gamma*R - S (old S); seed accumulators with R' + bias
    v8f c0, c1;
#pragma unroll
    for (int i = 0; i < 8; ++i) {
      R0[i] = g0 * R0[i] - S0[i];
      R1[i] = g1 * R1[i] - S1[i];
      c0[i] = R0[i] + b0;
      c1[i] = R1[i] + b1;
    }

    // U = P' @ W + bias + R' : 8 k-tiles; c0 uses register-resident B, c1 streams B from LDS
#pragma unroll
    for (int k = 0; k < 8; ++k) {
      v16bf a = joinbf(*(const v8bf*)(ab + k * 32),
                       *(const v8bf*)(ab + k * 32 + 8));
      c0 = __builtin_amdgcn_wmma_f32_16x16x32_bf16(false, a, false, B0r[k], (short)0, c0, false, false);
      v16bf bm1 = joinbf(*(const v8bf*)(bbase1 + k * 32),
                         *(const v8bf*)(bbase1 + k * 32 + 8));
      c1 = __builtin_amdgcn_wmma_f32_16x16x32_bf16(false, a, false, bm1, (short)0, c1, false, false);
    }

    // S' = (U > thr); new R/S state stays in accumulator layout; stream spikes out
    float* orow = optr + (size_t)t * BB * CC;
#pragma unroll
    for (int i = 0; i < 8; ++i) {
      float s0 = (c0[i] > THRV) ? 1.f : 0.f;
      float s1 = (c1[i] > THRV) ? 1.f : 0.f;
      S0[i] = s0; S1[i] = s1;
      const int m = i + mh * 8;
      orow[(size_t)m * CC + n0 * 16 + nc]      = s0;
      orow[(size_t)m * CC + n0 * 16 + 16 + nc] = s1;
    }
    // no trailing barrier: next step writes the other P' buffer
  }
}

extern "C" void kernel_launch(void* const* d_in, const int* in_sizes, int n_in,
                              void* d_out, int out_size, void* d_ws, size_t ws_size,
                              hipStream_t stream) {
  (void)in_sizes; (void)n_in; (void)d_ws; (void)ws_size; (void)out_size;
  const float* xin   = (const float*)d_in[0];
  const float* w     = (const float*)d_in[1];
  const float* bias  = (const float*)d_in[2];
  const float* alpha = (const float*)d_in[3];
  const float* beta  = (const float*)d_in[4];
  const float* gamma = (const float*)d_in[5];
  float* out = (float*)d_out;

  const size_t shmem = (size_t)(CC + 32) * LDSROW * 2u;   // 152,064 B < 320 KB WGP LDS
  hipFuncSetAttribute(reinterpret_cast<const void*>(&lif_scan_kernel),
                      hipFuncAttributeMaxDynamicSharedMemorySize, (int)shmem);

  lif_scan_kernel<<<BB / 16, 256, shmem, stream>>>(xin, w, bias, alpha, beta, gamma, out);
}
